// GCN_Layer_sage_16509854285892
// MI455X (gfx1250) — matrix operand order repaired and checked by reading, not compiled, not run on hardware
//
#include <hip/hip_runtime.h>
#include <hip/hip_bf16.h>

// GraphSAGE 3-layer forward for MI455X (gfx1250, wave32, WMMA).
// Strategy: build CSR (pull-style aggregation, no float atomics), then
// fused GEMM (out = agg@Wl^T + h@Wr^T + b [, relu]) using
// v_wmma_f32_16x16x32_bf16 with bf16 split-3 for ~fp32 accuracy.

#define GN 100000
#define GE 1600000
#define GD 128

typedef __attribute__((ext_vector_type(16))) __bf16 v16bf;
typedef __attribute__((ext_vector_type(8)))  float  v8f;

// ---------------------------------------------------------------- utilities
__global__ void zero_i32(int* p, int n) {
    int i = blockIdx.x * blockDim.x + threadIdx.x;
    if (i < n) p[i] = 0;
}

__global__ void copy_i32(const int* __restrict__ s, int* __restrict__ d, int n) {
    int i = blockIdx.x * blockDim.x + threadIdx.x;
    if (i < n) d[i] = s[i];
}

// ---------------------------------------------------------------- CSR build
__global__ void csr_count(const int* __restrict__ dst, int e, int* deg) {
    int i = blockIdx.x * blockDim.x + threadIdx.x;
    if (i < e) atomicAdd(&deg[dst[i]], 1);
}

// Single-block exclusive scan: row_start[0]=0, row_start[i+1]=sum deg[0..i].
__global__ void csr_scan(const int* __restrict__ deg, int* __restrict__ rs, int n) {
    __shared__ int buf[1024];
    __shared__ int carry;
    const int t = threadIdx.x;
    if (t == 0) { carry = 0; rs[0] = 0; }
    __syncthreads();
    for (int base = 0; base < n; base += 1024) {
        int v = (base + t < n) ? deg[base + t] : 0;
        buf[t] = v;
        __syncthreads();
        #pragma unroll
        for (int off = 1; off < 1024; off <<= 1) {
            int x = (t >= off) ? buf[t - off] : 0;
            __syncthreads();
            buf[t] += x;
            __syncthreads();
        }
        int incl = buf[t];
        if (base + t < n) rs[base + t + 1] = carry + incl;
        __syncthreads();
        if (t == 0) carry += buf[1023];
        __syncthreads();
    }
}

__global__ void csr_fill(const int* __restrict__ src, const int* __restrict__ dst,
                         int e, int* cursor, int* __restrict__ csr) {
    int i = blockIdx.x * blockDim.x + threadIdx.x;
    if (i < e) {
        int d = dst[i];
        int p = atomicAdd(&cursor[d], 1);
        csr[p] = src[i];
    }
}

// ------------------------------------------------------- mean aggregation
// One wave per node; lane owns 4 features (float4); neighbor indices are
// loaded 32-at-a-time and broadcast with __shfl (wave32).
__global__ void sage_aggregate(const float4* __restrict__ X,     // N x 32 float4
                               const int* __restrict__ rs,
                               const int* __restrict__ csr,
                               float4* __restrict__ Agg, int n) {
    const int lane = threadIdx.x & 31;
    const int node = blockIdx.x * 8 + (threadIdx.x >> 5);
    if (node >= n) return;
    const int s0 = rs[node];
    const int s1 = rs[node + 1];
    float4 acc = make_float4(0.f, 0.f, 0.f, 0.f);
    for (int j0 = s0; j0 < s1; j0 += 32) {
        int idx = 0;
        if (j0 + lane < s1) idx = csr[j0 + lane];
        int m = s1 - j0;
        if (m > 32) m = 32;
        for (int i = 0; i < m; ++i) {
            int s = __shfl(idx, i, 32);
            float4 v = X[(size_t)s * 32 + lane];
            acc.x += v.x; acc.y += v.y; acc.z += v.z; acc.w += v.w;
        }
    }
    const float inv = (s1 > s0) ? 1.0f / (float)(s1 - s0) : 0.0f;
    acc.x *= inv; acc.y *= inv; acc.z *= inv; acc.w *= inv;
    Agg[(size_t)node * 32 + lane] = acc;
}

// --------------------------------------------------------------- WMMA GEMM
__device__ __forceinline__ void split16(const float* __restrict__ f,
                                        v16bf& hi, v16bf& lo) {
    #pragma unroll
    for (int i = 0; i < 16; ++i) {
        float x = f[i];
        __bf16 h = (__bf16)x;
        hi[i] = h;
        lo[i] = (__bf16)(x - (float)h);
    }
}

// A tile (16x32, MxK): lane<16 -> M=lane, K chunks {c..c+7, c+16..c+23};
// lane>=16 -> M=lane-16, chunks shifted by 8 (ISA 7.12.2 16-bit A layout).
__device__ __forceinline__ void load_a(const float* __restrict__ row,
                                       int c0, float* __restrict__ a) {
    float4 p0 = *(const float4*)(row + c0);
    float4 p1 = *(const float4*)(row + c0 + 4);
    float4 p2 = *(const float4*)(row + c0 + 16);
    float4 p3 = *(const float4*)(row + c0 + 20);
    a[0]=p0.x; a[1]=p0.y; a[2]=p0.z; a[3]=p0.w;
    a[4]=p1.x; a[5]=p1.y; a[6]=p1.z; a[7]=p1.w;
    a[8]=p2.x; a[9]=p2.y; a[10]=p2.z; a[11]=p2.w;
    a[12]=p3.x; a[13]=p3.y; a[14]=p3.z; a[15]=p3.w;
}

// B tile (32x16, KxN): lane<16 -> N=lane, K chunks {c..c+7, c+16..c+23};
// lane>=16 -> N=lane-16, chunks shifted by 8 (matches A layout).
__device__ __forceinline__ void load_b(const float* __restrict__ p,
                                       float* __restrict__ b) {
    float4 p0 = *(const float4*)(p);
    float4 p1 = *(const float4*)(p + 4);
    float4 p2 = *(const float4*)(p + 16);
    float4 p3 = *(const float4*)(p + 20);
    b[0]=p0.x; b[1]=p0.y; b[2]=p0.z; b[3]=p0.w;
    b[4]=p1.x; b[5]=p1.y; b[6]=p1.z; b[7]=p1.w;
    b[8]=p2.x; b[9]=p2.y; b[10]=p2.z; b[11]=p2.w;
    b[12]=p3.x; b[13]=p3.y; b[14]=p3.z; b[15]=p3.w;
}

#define WMMA_BF16(Cacc, Aop, Bop) \
    Cacc = __builtin_amdgcn_wmma_f32_16x16x32_bf16(false, Aop, false, Bop, \
                                                   (short)0, Cacc, false, false)

// Out[m][n] = sum_k Agg[m][k]*Wl[n][k] + H[m][k]*Wr[n][k] + bias[n] (opt relu).
// grid.x = N/16 row blocks; (DOUT/16) waves per block, one 16x16 tile per wave.
template <int DOUT, bool RELU>
__global__ void sage_gemm(const float* __restrict__ Agg, const float* __restrict__ H,
                          const float* __restrict__ Wl,  const float* __restrict__ Wr,
                          const float* __restrict__ bias, float* __restrict__ Out) {
    const int lane = threadIdx.x & 31;
    const int nt   = threadIdx.x >> 5;    // N tile (column tile) for this wave
    const int rb   = blockIdx.x;          // 16-row block
    const int half = lane >> 4;
    const int lm   = lane & 15;

    const float* arow = Agg + (size_t)(rb * 16 + lm) * GD;
    const float* hrow = H   + (size_t)(rb * 16 + lm) * GD;
    const float* wlrow = Wl + (size_t)(nt * 16 + lm) * GD;
    const float* wrrow = Wr + (size_t)(nt * 16 + lm) * GD;

    v8f c = {0.f, 0.f, 0.f, 0.f, 0.f, 0.f, 0.f, 0.f};

    #pragma unroll
    for (int kt = 0; kt < GD / 32; ++kt) {
        const int c0 = kt * 32 + half * 8;    // A: first of two 8-float chunks
        const int kb = kt * 32 + half * 8;    // B: same two 8-float chunks as A
        float buf[16];
        v16bf ah, al, bh, bl;

        // neighbor path: Agg x Wl^T
        load_a(arow, c0, buf);  split16(buf, ah, al);
        load_b(wlrow + kb, buf); split16(buf, bh, bl);
        WMMA_BF16(c, ah, bh);
        WMMA_BF16(c, ah, bl);
        WMMA_BF16(c, al, bh);

        // root path: H x Wr^T
        load_a(hrow, c0, buf);  split16(buf, ah, al);
        load_b(wrrow + kb, buf); split16(buf, bh, bl);
        WMMA_BF16(c, ah, bh);
        WMMA_BF16(c, ah, bl);
        WMMA_BF16(c, al, bh);
    }

    const int col = nt * 16 + lm;
    const float bv = bias[col];
    #pragma unroll
    for (int v = 0; v < 8; ++v) {
        float r = c[v] + bv;
        if (RELU) r = fmaxf(r, 0.0f);
        // C layout: VGPR v -> row v (lanes 0-15) or 8+v (lanes 16-31)
        Out[(size_t)(rb * 16 + half * 8 + v) * DOUT + col] = r;
    }
}

// ---------------------------------------------------------------- launcher
extern "C" void kernel_launch(void* const* d_in, const int* in_sizes, int n_in,
                              void* d_out, int out_size, void* d_ws, size_t ws_size,
                              hipStream_t stream) {
    (void)in_sizes; (void)n_in; (void)out_size; (void)ws_size;

    const float* x    = (const float*)d_in[0];
    const int*   ei0  = (const int*)  d_in[1];   // (2,E): src=ei0, dst=ei0+E
    const int*   ei1  = (const int*)  d_in[2];
    const float* Wl1  = (const float*)d_in[3];
    const float* Wr1  = (const float*)d_in[4];
    const float* b1   = (const float*)d_in[5];
    const float* Wl2  = (const float*)d_in[6];
    const float* Wr2  = (const float*)d_in[7];
    const float* b2   = (const float*)d_in[8];
    const float* Wl3  = (const float*)d_in[9];
    const float* Wr3  = (const float*)d_in[10];
    const float* b3   = (const float*)d_in[11];
    float* out = (float*)d_out;

    // workspace carve-up (~168 MB)
    char*  w   = (char*)d_ws;
    size_t off = 0;
    auto take = [&](size_t bytes) -> void* {
        void* p = w + off;
        off = (off + bytes + 255) & ~(size_t)255;
        return p;
    };
    int*   deg    = (int*)take((size_t)GN * 4);
    int*   cursor = (int*)take((size_t)GN * 4);
    int*   rs1    = (int*)take((size_t)(GN + 1) * 4);
    int*   rs2    = (int*)take((size_t)(GN + 1) * 4);
    int*   csr1   = (int*)take((size_t)GE * 4);
    int*   csr2   = (int*)take((size_t)GE * 4);
    float* agg    = (float*)take((size_t)GN * GD * 4);
    float* h1     = (float*)take((size_t)GN * GD * 4);
    float* h2     = (float*)take((size_t)GN * GD * 4);

    const int TB = 256;
    const dim3 gE((GE + TB - 1) / TB);
    const dim3 gNn((GN + TB - 1) / TB);

    // ---- CSR for edge_index (layers 1 & 3)
    zero_i32<<<gNn, TB, 0, stream>>>(deg, GN);
    csr_count<<<gE, TB, 0, stream>>>(ei0 + GE, GE, deg);
    csr_scan<<<1, 1024, 0, stream>>>(deg, rs1, GN);
    copy_i32<<<gNn, TB, 0, stream>>>(rs1, cursor, GN);
    csr_fill<<<gE, TB, 0, stream>>>(ei0, ei0 + GE, GE, cursor, csr1);

    // ---- CSR for edge_idx_1_1 (layer 2)
    zero_i32<<<gNn, TB, 0, stream>>>(deg, GN);
    csr_count<<<gE, TB, 0, stream>>>(ei1 + GE, GE, deg);
    csr_scan<<<1, 1024, 0, stream>>>(deg, rs2, GN);
    copy_i32<<<gNn, TB, 0, stream>>>(rs2, cursor, GN);
    csr_fill<<<gE, TB, 0, stream>>>(ei1, ei1 + GE, GE, cursor, csr2);

    const dim3 gAgg(GN / 8);       // 8 waves/block, 1 wave per node
    const dim3 gGemm(GN / 16);     // 16-row blocks

    // ---- layer 1
    sage_aggregate<<<gAgg, TB, 0, stream>>>((const float4*)x, rs1, csr1,
                                            (float4*)agg, GN);
    sage_gemm<128, true><<<gGemm, 256, 0, stream>>>(agg, x, Wl1, Wr1, b1, h1);

    // ---- layer 2
    sage_aggregate<<<gAgg, TB, 0, stream>>>((const float4*)h1, rs2, csr2,
                                            (float4*)agg, GN);
    sage_gemm<128, true><<<gGemm, 256, 0, stream>>>(agg, h1, Wl2, Wr2, b2, h2);

    // ---- layer 3 (DOUT=64 -> 4 waves per block, no relu)
    sage_aggregate<<<gAgg, TB, 0, stream>>>((const float4*)h2, rs1, csr1,
                                            (float4*)agg, GN);
    sage_gemm<64, false><<<gGemm, 128, 0, stream>>>(agg, h2, Wl3, Wr3, b3, out);
}
